// QuantC2f_39084202394098
// MI455X (gfx1250) — matrix operand order repaired and checked by reading, not compiled, not run on hardware
//
#include <hip/hip_runtime.h>
#include <hip/hip_bf16.h>
#include <stddef.h>

// ---------------------------------------------------------------------------
// Quantized C2f block for MI455X (gfx1250, wave32, WMMA).
//
// All 6 convs run as implicit GEMMs on v_wmma_f32_16x16x32_bf16 with a
// double-buffered LDS pipeline using gfx1250 GLOBAL_LOAD_ASYNC_TO_LDS_B128
// (ASYNCcnt-tracked) when available, VGPR-bounce fallback otherwise.
// Activations between layers are the integer quant codes (0..255) stored as
// bf16 (exact); all scales (weight-quant scale * BN scale * input act scale)
// are folded into prepacked bf16 weight matrices; bias = folded BN shift.
//
// Workspace requirement: ~186 MB (xq 52MB + cat 105MB + tmp 26MB + weights).
// ---------------------------------------------------------------------------

typedef __attribute__((ext_vector_type(16))) __bf16 bf16x16;
typedef __attribute__((ext_vector_type(8)))  __bf16 bf16x8;
typedef __attribute__((ext_vector_type(8)))  float  floatx8;
typedef __attribute__((ext_vector_type(4)))  int    i32x4;

union FragAB { bf16x16 v; bf16x8 h[2]; };

#define HIMG 80
#define WIMG 80
#define HWPIX 6400          // 80*80
#define PIXELS 102400       // 16*6400
#define ROWP 72             // LDS row pitch (64 data + 8 pad) -> conflict-free

#if __has_builtin(__builtin_amdgcn_global_load_async_to_lds_b128)
#define ASYNC_LDS 1
#else
#define ASYNC_LDS 0
#endif

__device__ __forceinline__ void async_copy16(const void* g, void* l) {
#if ASYNC_LDS
  // prototype (leaked by clang diagnostic): arg0 = AS1 pointer to int4 vector
  __builtin_amdgcn_global_load_async_to_lds_b128(
      (__attribute__((address_space(1))) i32x4*)g,
      (__attribute__((address_space(3))) i32x4*)l, 0, 0);
#else
  (void)g; (void)l;
#endif
}

__device__ __forceinline__ void async_wait0() {
#if ASYNC_LDS
#if __has_builtin(__builtin_amdgcn_s_wait_asynccnt)
  __builtin_amdgcn_s_wait_asynccnt(0);
#else
  asm volatile("s_wait_asynccnt 0" ::: "memory");
#endif
#endif
}

// ---------------------------------------------------------------------------
// NCHW f32 -> NHWC bf16 (LDS-tiled transpose), 32 channels x 64 pixels / block
// ---------------------------------------------------------------------------
__global__ __launch_bounds__(256)
void nchw_to_nhwc_bf16(const float* __restrict__ x, __bf16* __restrict__ xq) {
  __shared__ float tile[32][65];
  const int pT = blockIdx.x * 64;
  const int cT = blockIdx.y * 32;
  const int t = threadIdx.x;

  const int px = t & 63, cl = t >> 6;
  const int p = pT + px;
  const int nI = p / HWPIX;
  const int sp = p - nI * HWPIX;
  const float* src = x + (size_t)nI * 256 * HWPIX + sp;
#pragma unroll
  for (int it = 0; it < 8; ++it) {
    const int c = cl + it * 4;
    tile[c][px] = src[(size_t)(cT + c) * HWPIX];
  }
  __syncthreads();
  const int ch = t & 31, pl = t >> 5;
#pragma unroll
  for (int it = 0; it < 8; ++it) {
    const int pp = pl + it * 8;
    xq[(size_t)(pT + pp) * 256 + cT + ch] = (__bf16)tile[ch][pp];
  }
}

// ---------------------------------------------------------------------------
// Per-output-channel weight prep: maxabs -> int8 fake-quant -> fold BN scale
// and per-input-channel activation scale; repack OIHW -> [o][t*Cin + i] bf16.
// ---------------------------------------------------------------------------
__global__ __launch_bounds__(256)
void prep_weights(const float* __restrict__ w,
                  const float* __restrict__ bn_g, const float* __restrict__ bn_b,
                  const float* __restrict__ bn_m, const float* __restrict__ bn_v,
                  const float* __restrict__ sIn, int Cin, int taps,
                  __bf16* __restrict__ Weff, float* __restrict__ bias) {
  const int o = blockIdx.x;
  const int K = Cin * taps;
  const float* wr = w + (size_t)o * K;

  __shared__ float red[256];
  float mx = 0.f;
  for (int e = threadIdx.x; e < K; e += 256) mx = fmaxf(mx, fabsf(wr[e]));
  red[threadIdx.x] = mx;
  __syncthreads();
  for (int s = 128; s > 0; s >>= 1) {
    if (threadIdx.x < s) red[threadIdx.x] = fmaxf(red[threadIdx.x], red[threadIdx.x + s]);
    __syncthreads();
  }
  const float scale = fmaxf(red[0] * (1.0f / 127.0f), 1e-8f);
  const float inv   = 1.0f / scale;
  const float bnS   = bn_g[o] * rsqrtf(bn_v[o] + 1e-5f);
  if (threadIdx.x == 0) bias[o] = bn_b[o] - bn_m[o] * bnS;

  __bf16* wd = Weff + (size_t)o * K;
  for (int e = threadIdx.x; e < K; e += 256) {
    const int i = e / taps;      // input channel (OIHW row = i*taps + t)
    const int t = e - i * taps;  // tap (ky*3+kx)
    float q = rintf(wr[e] * inv);
    q = fminf(fmaxf(q, -127.f), 127.f);
    const float sv = sIn ? fmaxf(sIn[i], 1e-8f) : 1.0f;
    wd[(size_t)t * Cin + i] = (__bf16)(q * scale * bnS * sv);
  }
}

__global__ void build_scat(const float* __restrict__ s1, const float* __restrict__ s2,
                           const float* __restrict__ s3, float* __restrict__ out) {
  const int i = blockIdx.x * blockDim.x + threadIdx.x;
  if (i < 256)      out[i] = s1[i];
  else if (i < 384) out[i] = s2[i - 256];
  else if (i < 512) out[i] = s3[i - 384];
}

// ---------------------------------------------------------------------------
// Implicit-GEMM conv. Block: 256 thr (8 waves), tile M=128 x N=128.
// Wave tile 32x64 -> 2x4 WMMA accumulators. K staged 64 at a time through a
// double-buffered LDS pipeline; 16 v_wmma per barrier. taps=1 or 9 (shifted
// GEMM taps, zero-fill at image border). Epilogue fuses BN bias + ReLU +
// uint8 fake-quant; bf16 codes (intermediate) or q*s f32 NCHW (final).
// ---------------------------------------------------------------------------
__global__ __launch_bounds__(256)
void conv_wmma(const __bf16* __restrict__ A, int aStride,
               const __bf16* __restrict__ W, const float* __restrict__ bias,
               const float* __restrict__ actS, int Cin, int taps,
               __bf16* __restrict__ outB, int outStride, int outBase,
               float* __restrict__ outF, int finalMode) {
  __shared__ __align__(16) __bf16 sA[2][128 * ROWP];
  __shared__ __align__(16) __bf16 sB[2][128 * ROWP];

  const int tid  = threadIdx.x;
  const int wave = tid >> 5, lane = tid & 31;
  const int wm = wave & 3;          // 4 wave rows  -> M offset 32 each
  const int wn = wave >> 2;         // 2 wave cols  -> N offset 64 each
  const int lo = lane & 15, hi = lane >> 4;
  const int mBase = blockIdx.x * 128;
  const int nBase = blockIdx.y * 128;

  // staging role: one (row, 32-element half) = 64B per thread per matrix
  const int sRow  = tid >> 1;       // 0..127
  const int sHalf = tid & 1;        // 0..1

  const int pRow = mBase + sRow;
  const int nImg = pRow / HWPIX;
  const int sp   = pRow - nImg * HWPIX;
  const int py   = sp / WIMG;
  const int px   = sp - py * WIMG;

  floatx8 acc[2][4];
#pragma unroll
  for (int i = 0; i < 2; ++i)
#pragma unroll
    for (int j = 0; j < 4; ++j)
#pragma unroll
      for (int k = 0; k < 8; ++k) acc[i][j][k] = 0.f;

  const int kChunks = Cin >> 6;     // 64-wide K chunks
  const int total   = taps * kChunks;
  const int KW      = taps * Cin;

  uint4 ra[4], rb[4];               // sync-path staging registers

  // Issue chunk (t,kc) into LDS buffer `buf`. Async path writes LDS directly;
  // sync path parks data in ra/rb for commitStage.
  auto issueStage = [&](int t, int kc, int buf) {
    int dy = 0, dx = 0;
    if (taps == 9) { const int ky = t / 3; dy = ky - 1; dx = t - ky * 3 - 1; }
    const int yy = py + dy, xx = px + dx;
    const bool inb = ((unsigned)yy < (unsigned)HIMG) && ((unsigned)xx < (unsigned)WIMG);
    const __bf16* aSrc = A + (size_t)(nImg * HWPIX + yy * WIMG + xx) * aStride
                         + kc * 64 + sHalf * 32;
    const __bf16* wSrc = W + (size_t)(nBase + sRow) * KW + t * Cin + kc * 64 + sHalf * 32;
    __bf16* aDst = &sA[buf][sRow * ROWP + sHalf * 32];
    __bf16* bDst = &sB[buf][sRow * ROWP + sHalf * 32];
#if ASYNC_LDS
    if (inb) {
#pragma unroll
      for (int j = 0; j < 4; ++j) async_copy16(aSrc + j * 8, aDst + j * 8);
    } else {
      const uint4 z = {0, 0, 0, 0};
#pragma unroll
      for (int j = 0; j < 4; ++j) ((uint4*)aDst)[j] = z;
    }
#pragma unroll
    for (int j = 0; j < 4; ++j) async_copy16(wSrc + j * 8, bDst + j * 8);
#else
    const uint4 z = {0, 0, 0, 0};
#pragma unroll
    for (int j = 0; j < 4; ++j) ra[j] = inb ? ((const uint4*)aSrc)[j] : z;
#pragma unroll
    for (int j = 0; j < 4; ++j) rb[j] = ((const uint4*)wSrc)[j];
    (void)aDst; (void)bDst;
#endif
  };

  auto commitStage = [&](int buf) {
#if !ASYNC_LDS
    __bf16* aDst = &sA[buf][sRow * ROWP + sHalf * 32];
    __bf16* bDst = &sB[buf][sRow * ROWP + sHalf * 32];
#pragma unroll
    for (int j = 0; j < 4; ++j) ((uint4*)aDst)[j] = ra[j];
#pragma unroll
    for (int j = 0; j < 4; ++j) ((uint4*)bDst)[j] = rb[j];
#else
    (void)buf;
#endif
  };

  issueStage(0, 0, 0);
  commitStage(0);

  int tc = 0, kcc = 0;
  for (int ci = 0; ci < total; ++ci) {
    const int cur = ci & 1;
    async_wait0();
    __syncthreads();                     // chunk ci resident in buf[cur]

    int tn = tc, kn = kcc + 1;
    if (kn == kChunks) { kn = 0; ++tn; }
    const bool more = (ci + 1) < total;
    if (more) issueStage(tn, kn, cur ^ 1);  // overlap next chunk with compute
    tc = tn; kcc = kn;

    // ---- compute chunk ci: 2 K-steps x (2 M x 4 N) WMMAs ----
#pragma unroll
    for (int ks = 0; ks < 2; ++ks) {
      FragAB a[2];
#pragma unroll
      for (int mt = 0; mt < 2; ++mt) {
        // A layout (ISA 7.12.2): lane=row; lanes 0-15 K 0..7 & 16..23,
        // lanes 16-31 K 8..15 & 24..31.
        const __bf16* pa = &sA[cur][(wm * 32 + mt * 16 + lo) * ROWP + ks * 32 + hi * 8];
        a[mt].h[0] = *(const bf16x8*)pa;
        a[mt].h[1] = *(const bf16x8*)(pa + 16);
      }
      FragAB b[4];
#pragma unroll
      for (int nt = 0; nt < 4; ++nt) {
        // B layout: lane=col; lanes 0-15: K 0..7 & 16..23, lanes 16-31: K 8..15 & 24..31.
        const __bf16* pb = &sB[cur][(wn * 64 + nt * 16 + lo) * ROWP + ks * 32 + hi * 8];
        b[nt].h[0] = *(const bf16x8*)pb;
        b[nt].h[1] = *(const bf16x8*)(pb + 16);
      }
#pragma unroll
      for (int mt = 0; mt < 2; ++mt)
#pragma unroll
        for (int nt = 0; nt < 4; ++nt)
          acc[mt][nt] = __builtin_amdgcn_wmma_f32_16x16x32_bf16(
              false, a[mt].v, false, b[nt].v, (short)0, acc[mt][nt], false, false);
    }

    if (more) commitStage(cur ^ 1);      // sync path: LDS store behind compute
  }

  // ----- epilogue: BN bias + ReLU + uint8 fake-quant -----
#pragma unroll
  for (int nt = 0; nt < 4; ++nt) {
    const int ncol = nBase + wn * 64 + nt * 16 + lo;
    const float bv = bias[ncol];
    const float s = fmaxf(actS[ncol], 1e-8f);
    const float invs = 1.0f / s;
#pragma unroll
    for (int mt = 0; mt < 2; ++mt) {
      const int m0 = mBase + wm * 32 + mt * 16 + hi * 8;  // C/D: M = hi*8 + r
      if (!finalMode) {
#pragma unroll
        for (int r = 0; r < 8; ++r) {
          float y = acc[mt][nt][r] + bv;
          y = fmaxf(y, 0.f);
          const float q = fminf(rintf(y * invs), 255.f);
          outB[(size_t)(m0 + r) * outStride + outBase + ncol] = (__bf16)q;
        }
      } else {
        float vals[8];
#pragma unroll
        for (int r = 0; r < 8; ++r) {
          float y = acc[mt][nt][r] + bv;
          y = fmaxf(y, 0.f);
          const float q = fminf(rintf(y * invs), 255.f);
          vals[r] = q * s;
        }
        const int nI = m0 / HWPIX;
        const int s0 = m0 - nI * HWPIX;
        float* dst = outF + (size_t)(nI * 256 + ncol) * HWPIX + s0;
        float4 v0; v0.x = vals[0]; v0.y = vals[1]; v0.z = vals[2]; v0.w = vals[3];
        float4 v1; v1.x = vals[4]; v1.y = vals[5]; v1.z = vals[6]; v1.w = vals[7];
        ((float4*)dst)[0] = v0;
        ((float4*)dst)[1] = v1;
      }
    }
  }
}

// ---------------------------------------------------------------------------
// Host launcher
// ---------------------------------------------------------------------------
extern "C" void kernel_launch(void* const* d_in, const int* in_sizes, int n_in,
                              void* d_out, int out_size, void* d_ws, size_t ws_size,
                              hipStream_t stream) {
  (void)in_sizes; (void)n_in; (void)out_size; (void)ws_size;

  // Input order (setup_inputs insertion order, depth-first):
  // 0:x  | cv1: 1:w 2:g 3:b 4:m 5:v 6:act_s | cv2: 7..12
  // m0.cv1: 13..18 | m0.cv2: 19..24 | m1.cv1: 25..30 | m1.cv2: 31..36
  const float* X = (const float*)d_in[0];
  auto F = [&](int i) { return (const float*)d_in[i]; };

  char* ws = (char*)d_ws;
  size_t off = 0;
  auto alloc = [&](size_t bytes) {
    void* p = ws + off;
    off += (bytes + 255) & ~(size_t)255;
    return p;
  };

  __bf16* xq   = (__bf16*)alloc((size_t)PIXELS * 256 * 2);   // x as NHWC bf16
  __bf16* cat  = (__bf16*)alloc((size_t)PIXELS * 512 * 2);   // fused concat buffer
  __bf16* tbuf = (__bf16*)alloc((size_t)PIXELS * 128 * 2);   // bottleneck temp
  __bf16* W1   = (__bf16*)alloc((size_t)256 * 256 * 2);
  __bf16* W2   = (__bf16*)alloc((size_t)256 * 512 * 2);
  __bf16* Wm0a = (__bf16*)alloc((size_t)128 * 1152 * 2);
  __bf16* Wm0b = (__bf16*)alloc((size_t)128 * 1152 * 2);
  __bf16* Wm1a = (__bf16*)alloc((size_t)128 * 1152 * 2);
  __bf16* Wm1b = (__bf16*)alloc((size_t)128 * 1152 * 2);
  float* b1   = (float*)alloc(256 * 4);
  float* b2   = (float*)alloc(256 * 4);
  float* bm0a = (float*)alloc(128 * 4);
  float* bm0b = (float*)alloc(128 * 4);
  float* bm1a = (float*)alloc(128 * 4);
  float* bm1b = (float*)alloc(128 * 4);
  float* sCat = (float*)alloc(512 * 4);

  nchw_to_nhwc_bf16<<<dim3(PIXELS / 64, 256 / 32), 256, 0, stream>>>(X, xq);

  prep_weights<<<256, 256, 0, stream>>>(F(1), F(2), F(3), F(4), F(5), nullptr,
                                        256, 1, W1, b1);
  prep_weights<<<128, 256, 0, stream>>>(F(13), F(14), F(15), F(16), F(17), F(6) + 128,
                                        128, 9, Wm0a, bm0a);
  prep_weights<<<128, 256, 0, stream>>>(F(19), F(20), F(21), F(22), F(23), F(18),
                                        128, 9, Wm0b, bm0b);
  prep_weights<<<128, 256, 0, stream>>>(F(25), F(26), F(27), F(28), F(29), F(24),
                                        128, 9, Wm1a, bm1a);
  prep_weights<<<128, 256, 0, stream>>>(F(31), F(32), F(33), F(34), F(35), F(30),
                                        128, 9, Wm1b, bm1b);
  build_scat<<<2, 256, 0, stream>>>(F(6), F(24), F(36), sCat);
  prep_weights<<<256, 256, 0, stream>>>(F(7), F(8), F(9), F(10), F(11), sCat,
                                        512, 1, W2, b2);

  const int MB = PIXELS / 128;  // 800 M-blocks

  conv_wmma<<<dim3(MB, 2), 256, 0, stream>>>(xq, 256, W1, b1, F(6), 256, 1,
                                             cat, 512, 0, nullptr, 0);
  conv_wmma<<<dim3(MB, 1), 256, 0, stream>>>(cat + 128, 512, Wm0a, bm0a, F(18), 128, 9,
                                             tbuf, 128, 0, nullptr, 0);
  conv_wmma<<<dim3(MB, 1), 256, 0, stream>>>(tbuf, 128, Wm0b, bm0b, F(24), 128, 9,
                                             cat, 512, 256, nullptr, 0);
  conv_wmma<<<dim3(MB, 1), 256, 0, stream>>>(cat + 256, 512, Wm1a, bm1a, F(30), 128, 9,
                                             tbuf, 128, 0, nullptr, 0);
  conv_wmma<<<dim3(MB, 1), 256, 0, stream>>>(tbuf, 128, Wm1b, bm1b, F(36), 128, 9,
                                             cat, 512, 384, nullptr, 0);
  conv_wmma<<<dim3(MB, 2), 256, 0, stream>>>(cat, 512, W2, b2, F(12), 512, 1,
                                             nullptr, 0, 0, (float*)d_out, 1);
}
